// SNNBPTT3_37031208026550
// MI455X (gfx1250) — matrix operand order repaired and checked by reading, not compiled, not run on hardware
//
#include <hip/hip_runtime.h>
#include <hip/hip_bf16.h>
#include <math.h>

// ---------------------------------------------------------------------------
// SNN forward (snntorch Leaky, forward only):
//   cur1 = x @ W1^T + b1                        (512x784 @ 784x2048)
//   100 LIF steps, cur2 = spk1 @ W2^T + b2 with binary spikes
//   out  = softmax(mean_t spk2)                 (512x10, f32)
// MI455X strategy: bf16 WMMA 16x16x32 both GEMMs, fp32 accumulate; all
// recurrent state on-chip (VGPRs + LDS); guard-free b128 global loads.
// ---------------------------------------------------------------------------

typedef __attribute__((ext_vector_type(16))) __bf16 v16bf;
typedef __attribute__((ext_vector_type(8)))  float  v8f;

union Frag {
    v16bf        v;
    unsigned int u[8];
};

__device__ __forceinline__ unsigned int pack_bf(float lo, float hi) {
    union { __bf16 b[2]; unsigned int u; } t;
    t.b[0] = (__bf16)lo;          // native f32->bf16 cvt (RNE) on gfx1250
    t.b[1] = (__bf16)hi;
    return t.u;
}

#define B_SZ   512
#define D_IN   784
#define D_H    2048
#define D_OUT  10
#define NSTEP  100
#define ROWS   16           // batch rows per workgroup (WMMA M)
#define T2     512          // threads in kernel 2 (16 waves)
#define SPK_STRIDE 2056     // 2048 + 8B pad: avoids LDS bank conflicts

// ---------------------------------------------------------------------------
// Kernel 1: cur1[512][2048] = x[512][784] @ W1^T + b1   (bf16 WMMA, f32 acc)
// One wave per 16(M) x 64(N): A fragment reused across 4 N-tiles.
// grid = (32, 32), block = 256 (8 waves). K: 24 clean iters + static tail.
// ---------------------------------------------------------------------------
__global__ void __launch_bounds__(256) snn_cur1_gemm(
    const float* __restrict__ x, const float* __restrict__ W1,
    const float* __restrict__ b1, float* __restrict__ cur1)
{
    const int lane = threadIdx.x & 31;
    const int wave = threadIdx.x >> 5;
    const int half = lane >> 4;          // 0: lanes 0-15, 1: lanes 16-31
    const int r    = lane & 15;
    const int m0   = blockIdx.y * 16;                    // 0..496
    const int n0   = (blockIdx.x * 8 + wave) * 64;       // 0..16320

    const float* __restrict__ xrow = x + (size_t)(m0 + r) * D_IN;
    const float* __restrict__ w1p[4];
    #pragma unroll
    for (int t = 0; t < 4; ++t)
        w1p[t] = W1 + (size_t)(n0 + t * 16 + r) * D_IN;

    v8f acc[4] = {v8f{}, v8f{}, v8f{}, v8f{}};

    // ---- 24 guard-free K iterations (k <= 767 < 784) ----------------------
    #pragma unroll 2
    for (int kb = 0; kb < 768; kb += 32) {
        // A fragment (16x32 bf16): lane half0 K {0..7,16..23}, half1 +8
        Frag a;
        {
            const int k0 = kb + half * 8;
            float4 a0 = *(const float4*)(xrow + k0);
            float4 a1 = *(const float4*)(xrow + k0 + 4);
            float4 a2 = *(const float4*)(xrow + k0 + 16);
            float4 a3 = *(const float4*)(xrow + k0 + 20);
            a.u[0] = pack_bf(a0.x, a0.y);  a.u[1] = pack_bf(a0.z, a0.w);
            a.u[2] = pack_bf(a1.x, a1.y);  a.u[3] = pack_bf(a1.z, a1.w);
            a.u[4] = pack_bf(a2.x, a2.y);  a.u[5] = pack_bf(a2.z, a2.w);
            a.u[6] = pack_bf(a3.x, a3.y);  a.u[7] = pack_bf(a3.z, a3.w);
        }
        // B fragments (32x16 bf16): lane half0 K {0..7,16..23}, half1 +8
        #pragma unroll
        for (int t = 0; t < 4; ++t) {
            const int k0 = kb + half * 8;
            float4 b0 = *(const float4*)(w1p[t] + k0);
            float4 b1v = *(const float4*)(w1p[t] + k0 + 4);
            float4 b2v = *(const float4*)(w1p[t] + k0 + 16);
            float4 b3 = *(const float4*)(w1p[t] + k0 + 20);
            Frag b;
            b.u[0] = pack_bf(b0.x, b0.y);   b.u[1] = pack_bf(b0.z, b0.w);
            b.u[2] = pack_bf(b1v.x, b1v.y); b.u[3] = pack_bf(b1v.z, b1v.w);
            b.u[4] = pack_bf(b2v.x, b2v.y); b.u[5] = pack_bf(b2v.z, b2v.w);
            b.u[6] = pack_bf(b3.x, b3.y);   b.u[7] = pack_bf(b3.z, b3.w);
            acc[t] = __builtin_amdgcn_wmma_f32_16x16x32_bf16(
                         false, a.v, false, b.v, (short)0, acc[t], false, false);
        }
    }

    // ---- static tail: kb = 768, valid k = 768..783 -------------------------
    {
        Frag a;
        {   // A: K slots 0..7 (k0..k0+7) valid for both halves; 16..31 -> 0
            const int k0 = 768 + half * 8;
            float4 a0 = *(const float4*)(xrow + k0);
            float4 a1 = *(const float4*)(xrow + k0 + 4);
            a.u[0] = pack_bf(a0.x, a0.y);  a.u[1] = pack_bf(a0.z, a0.w);
            a.u[2] = pack_bf(a1.x, a1.y);  a.u[3] = pack_bf(a1.z, a1.w);
            a.u[4] = 0; a.u[5] = 0; a.u[6] = 0; a.u[7] = 0;
        }
        #pragma unroll
        for (int t = 0; t < 4; ++t) {
            Frag b;
            {   // B: K slots 0..7 (k0..k0+7) valid for both halves; 16..31 -> 0
                const int k0 = 768 + half * 8;
                float4 b0 = *(const float4*)(w1p[t] + k0);
                float4 b1v = *(const float4*)(w1p[t] + k0 + 4);
                b.u[0] = pack_bf(b0.x, b0.y);   b.u[1] = pack_bf(b0.z, b0.w);
                b.u[2] = pack_bf(b1v.x, b1v.y); b.u[3] = pack_bf(b1v.z, b1v.w);
                b.u[4] = 0; b.u[5] = 0; b.u[6] = 0; b.u[7] = 0;
            }
            acc[t] = __builtin_amdgcn_wmma_f32_16x16x32_bf16(
                         false, a.v, false, b.v, (short)0, acc[t], false, false);
        }
    }

    // C layout: VGPR v, lanes 0-15 -> (M=v, N=lane); lanes 16-31 -> (M=v+8)
    #pragma unroll
    for (int t = 0; t < 4; ++t) {
        const int col  = n0 + t * 16 + r;
        const float bb = b1[col];
        #pragma unroll
        for (int v = 0; v < 8; ++v)
            cur1[(size_t)(m0 + v + 8 * half) * D_H + col] = acc[t][v] + bb;
    }
}

// ---------------------------------------------------------------------------
// Kernel 2: 100-step LIF recurrence + softmax. grid = 32, block = 512.
// ---------------------------------------------------------------------------
__global__ void __launch_bounds__(T2) snn_time_loop(
    const float* __restrict__ cur1_ws, const float* __restrict__ W2,
    const float* __restrict__ b2, const float* __restrict__ beta1p,
    const float* __restrict__ beta2p, float* __restrict__ out)
{
    __shared__ unsigned char spkLDS[ROWS * SPK_STRIDE];   // ~32.1 KB spikes
    __shared__ float redLDS[16 * 256];                    // 16 KB wave partials
    __shared__ float smLDS[ROWS * D_OUT];                 // softmax staging

    const int tid  = threadIdx.x;
    const int lane = tid & 31;
    const int wave = tid >> 5;
    const int half = lane >> 4;
    const int r    = lane & 15;
    const int r0   = blockIdx.x * ROWS;                   // global batch base

    const float beta1 = fminf(fmaxf(beta1p[0], 0.0f), 1.0f);
    const float beta2 = fminf(fmaxf(beta2p[0], 0.0f), 1.0f);

    // Thread-private slice of mem1/cur1: row (tid>>5), 64 consecutive h.
    const int myrow = tid >> 5;                // 0..15
    const int hbase = (tid & 31) << 6;         // 0..1984
    float mem1[64];
    float cur1r[64];
    {
        const float4* src =
            (const float4*)(cur1_ws + (size_t)(r0 + myrow) * D_H + hbase);
        #pragma unroll
        for (int q = 0; q < 16; ++q) {
            float4 f = src[q];
            cur1r[4 * q + 0] = f.x; cur1r[4 * q + 1] = f.y;
            cur1r[4 * q + 2] = f.z; cur1r[4 * q + 3] = f.w;
            mem1[4 * q + 0] = 0.0f; mem1[4 * q + 1] = 0.0f;
            mem1[4 * q + 2] = 0.0f; mem1[4 * q + 3] = 0.0f;
        }
    }

    // Loop-invariant W2 B-fragments: wave w owns K-chunks 4w..4w+3.
    // B[k][n] = W2[n][k]; cols 10..15 zero-padded.
    Frag bfr[4];
    #pragma unroll
    for (int c = 0; c < 4; ++c) {
        const int kb = (wave * 4 + c) * 32;
        #pragma unroll
        for (int v = 0; v < 8; ++v) {
            int k = kb + ((v < 4) ? 2 * v : 16 + 2 * (v - 4)) + half * 8;
            float f0 = (r < D_OUT) ? W2[r * D_H + k]     : 0.0f;
            float f1 = (r < D_OUT) ? W2[r * D_H + k + 1] : 0.0f;
            bfr[c].u[v] = pack_bf(f0, f1);
        }
    }

    // Layer-2 state: one thread per (row, o) for tid < 160.
    const int  row2   = tid / D_OUT;
    const int  o2     = tid - row2 * D_OUT;
    const bool owns2  = (tid < ROWS * D_OUT);
    float mem2    = 0.0f;
    float spk2sum = 0.0f;
    const float bias2 = owns2 ? b2[o2] : 0.0f;

    for (int t = 0; t < NSTEP; ++t) {
        // ---- layer-1 LIF update; spikes packed 4/uint into LDS ------------
        #pragma unroll
        for (int q = 0; q < 16; ++q) {
            unsigned int pk = 0;
            #pragma unroll
            for (int b = 0; b < 4; ++b) {
                int   j     = q * 4 + b;
                float m     = mem1[j];
                float reset = (m > 1.0f) ? 1.0f : 0.0f;   // uses mem_{t-1}
                m = beta1 * m + cur1r[j] - reset;
                mem1[j] = m;
                pk |= ((m > 1.0f) ? 1u : 0u) << (8 * b);  // spk = H(m - thr)
            }
            *(unsigned int*)&spkLDS[myrow * SPK_STRIDE + hbase + q * 4] = pk;
        }
        __syncthreads();

        // ---- cur2 partial via WMMA: A = spikes (bf16 0/1), B = W2 frags ---
        v8f acc = {};
        #pragma unroll
        for (int c = 0; c < 4; ++c) {
            const int kb = (wave * 4 + c) * 32;
            Frag a;
            #pragma unroll
            for (int v = 0; v < 8; ++v) {
                int k = kb + ((v < 4) ? 2 * v : 16 + 2 * (v - 4)) + (half ? 8 : 0);
                unsigned short s =
                    *(const unsigned short*)(spkLDS + r * SPK_STRIDE + k);
                unsigned int lo = (s & 0x0001u) ? 0x3F80u : 0u;   // bf16 1.0
                unsigned int hi = (s & 0x0100u) ? 0x3F80u : 0u;
                a.u[v] = lo | (hi << 16);
            }
            acc = __builtin_amdgcn_wmma_f32_16x16x32_bf16(
                      false, a.v, false, bfr[c].v, (short)0, acc, false, false);
        }
        #pragma unroll
        for (int v = 0; v < 8; ++v)
            redLDS[wave * 256 + (v + 8 * half) * 16 + r] = acc[v];
        __syncthreads();

        // ---- cross-wave reduce + layer-2 LIF ------------------------------
        if (owns2) {
            float s = bias2;
            #pragma unroll
            for (int w = 0; w < 16; ++w)
                s += redLDS[w * 256 + row2 * 16 + o2];
            float reset = (mem2 > 1.0f) ? 1.0f : 0.0f;
            mem2 = beta2 * mem2 + s - reset;
            spk2sum += (mem2 > 1.0f) ? 1.0f : 0.0f;
        }
        __syncthreads();   // protect spkLDS/redLDS for next step
    }

    // ---- mean over steps + softmax per batch row --------------------------
    if (owns2) smLDS[row2 * D_OUT + o2] = spk2sum * (1.0f / NSTEP);
    __syncthreads();
    if (tid < ROWS) {
        float v[D_OUT], mx = -1e30f, s = 0.0f;
        #pragma unroll
        for (int o = 0; o < D_OUT; ++o) {
            v[o] = smLDS[tid * D_OUT + o];
            mx = fmaxf(mx, v[o]);
        }
        #pragma unroll
        for (int o = 0; o < D_OUT; ++o) { v[o] = __expf(v[o] - mx); s += v[o]; }
        float inv = 1.0f / s;
        #pragma unroll
        for (int o = 0; o < D_OUT; ++o)
            out[(size_t)(r0 + tid) * D_OUT + o] = v[o] * inv;
    }
}

// ---------------------------------------------------------------------------
extern "C" void kernel_launch(void* const* d_in, const int* in_sizes, int n_in,
                              void* d_out, int out_size, void* d_ws, size_t ws_size,
                              hipStream_t stream) {
    const float* x     = (const float*)d_in[0];   // [512,784]
    const float* W1    = (const float*)d_in[1];   // [2048,784]
    const float* b1    = (const float*)d_in[2];   // [2048]
    const float* W2    = (const float*)d_in[3];   // [10,2048]
    const float* b2    = (const float*)d_in[4];   // [10]
    const float* beta1 = (const float*)d_in[5];   // scalar
    const float* beta2 = (const float*)d_in[6];   // scalar
    float* out   = (float*)d_out;                 // [512,10]
    float* cur1  = (float*)d_ws;                  // [512,2048] f32 = 4 MB

    dim3 g1(D_H / (16 * 64 / 16), B_SZ / 16);     // (32, 32)
    snn_cur1_gemm<<<g1, 256, 0, stream>>>(x, W1, b1, cur1);

    snn_time_loop<<<B_SZ / ROWS, T2, 0, stream>>>(cur1, W2, b2, beta1, beta2, out);
}
